// SelfAttentionBlock_72052371358195
// MI455X (gfx1250) — matrix operand order repaired and checked by reading, not compiled, not run on hardware
//
#include <hip/hip_runtime.h>

// ---------------- problem constants ----------------
#define Bb   4
#define Ss   4096
#define Cc   2048
#define Oo   2048
#define Hh   8
#define DHd  256
#define MTOK (Bb * Ss)   // 16384 rows
#define EPSl 1e-5f

typedef __bf16 bf16;
typedef __attribute__((ext_vector_type(16))) __bf16 v16bf;
typedef __attribute__((ext_vector_type(8)))  __bf16 v8bf;
typedef __attribute__((ext_vector_type(8)))  float  v8f;

// ---------------- bf16 helpers (bit-exact, no native-cvt dependence) -------
__device__ __forceinline__ float bf2f(bf16 b) {
    unsigned short u = __builtin_bit_cast(unsigned short, b);
    unsigned int   w = ((unsigned int)u) << 16;
    return __builtin_bit_cast(float, w);
}
__device__ __forceinline__ bf16 f2bf(float f) {
    unsigned int w = __builtin_bit_cast(unsigned int, f);
    unsigned int r = w + 0x7FFFu + ((w >> 16) & 1u);   // round-nearest-even
    unsigned short u = (unsigned short)(r >> 16);
    return __builtin_bit_cast(bf16, u);
}

// ---------------- elementwise f32 -> bf16 ----------------
__global__ __launch_bounds__(256)
void f32_to_bf16_kernel(const float* __restrict__ in, bf16* __restrict__ out, int n) {
    int i = blockIdx.x * 256 + threadIdx.x;
    if (i < n) out[i] = f2bf(in[i]);
}

// ---------------- tiled transpose + convert: W[rows,cols] -> Wt[cols,rows] bf16
__global__ __launch_bounds__(256)
void transpose_f32_to_bf16(const float* __restrict__ W, bf16* __restrict__ Wt,
                           int rows, int cols) {
    __shared__ float tile[32][33];
    const int c0 = blockIdx.x * 32;
    const int r0 = blockIdx.y * 32;
    const int tx = threadIdx.x;      // 0..31
    const int ty = threadIdx.y;      // 0..7
    #pragma unroll
    for (int j = 0; j < 32; j += 8)
        tile[ty + j][tx] = W[(size_t)(r0 + ty + j) * cols + (c0 + tx)];
    __syncthreads();
    #pragma unroll
    for (int j = 0; j < 32; j += 8)
        Wt[(size_t)(c0 + ty + j) * rows + (r0 + tx)] = f2bf(tile[tx][ty + j]);
}

// ---------------- WMMA bf16 GEMM ----------------
// D[M,N] = A[M,K] * Bt[N,K]^T (+bias) (+resid)
// MODE 0: write bf16 (QKV projections).  MODE 1: write f32 with residual add.
// Block = 128 threads (4 waves, 2x2 of 64x64 wave tiles) -> 128x128 block tile.
template <int MODE>
__global__ __launch_bounds__(128)
void gemm_wmma_bf16(const bf16* __restrict__ A,     // [M,K] row-major
                    const bf16* __restrict__ Bt,    // [N,K] row-major (B transposed)
                    const float* __restrict__ bias, // [N]
                    const float* __restrict__ resid,// [M,N] (MODE 1) or nullptr
                    bf16*  __restrict__ outb,       // [M,N] (MODE 0)
                    float* __restrict__ outf,       // [M,N] (MODE 1)
                    int M, int N, int K)
{
    const int lane = threadIdx.x & 31;
    const int wave = threadIdx.x >> 5;
    const int r    = lane & 15;      // row-within-tile (A) / column-within-tile (B, C/D)
    const int hi   = lane >> 4;      // K-half selector (A/B), M-half selector (C/D)

    const int m0 = blockIdx.y * 128 + (wave >> 1) * 64;
    const int n0 = blockIdx.x * 128 + (wave & 1) * 64;

    v8f acc[4][4];
    #pragma unroll
    for (int i = 0; i < 4; i++)
        #pragma unroll
        for (int j = 0; j < 4; j++)
            acc[i][j] = {};

    const bf16* Arow[4];
    const bf16* Brow[4];
    #pragma unroll
    for (int i = 0; i < 4; i++) Arow[i] = A  + (size_t)(m0 + i * 16 + r) * K;
    #pragma unroll
    for (int j = 0; j < 4; j++) Brow[j] = Bt + (size_t)(n0 + j * 16 + r) * K;

    for (int k0 = 0; k0 < K; k0 += 32) {
        v16bf a[4], b[4];
        // A 16x32 fragment: lane holds row r; K = hi*16 .. hi*16+15 (contiguous)
        #pragma unroll
        for (int i = 0; i < 4; i++) {
            a[i] = *(const v16bf*)(Arow[i] + k0 + hi * 16);
        }
        // B 32x16 fragment: lane holds column r; K = hi*16 .. hi*16+15 (contiguous in Bt)
        #pragma unroll
        for (int j = 0; j < 4; j++)
            b[j] = *(const v16bf*)(Brow[j] + k0 + hi * 16);

        #pragma unroll
        for (int i = 0; i < 4; i++)
            #pragma unroll
            for (int j = 0; j < 4; j++)
                acc[i][j] = __builtin_amdgcn_wmma_f32_16x16x32_bf16(
                    false, a[i], false, b[j], (short)0, acc[i][j], false, false);
    }

    // Epilogue: C/D layout — VGPR t holds (M = hi*8 + t, N = r) within each 16x16 tile
    #pragma unroll
    for (int j = 0; j < 4; j++) {
        const int col = n0 + j * 16 + r;
        const float bj = bias[col];
        #pragma unroll
        for (int i = 0; i < 4; i++) {
            const int row0 = m0 + i * 16 + hi * 8;
            #pragma unroll
            for (int t = 0; t < 8; t++) {
                const size_t idx = (size_t)(row0 + t) * N + col;
                float v = acc[i][j][t] + bj;
                if (MODE == 0) {
                    outb[idx] = f2bf(v);
                } else {
                    outf[idx] = v + resid[idx];
                }
            }
        }
    }
}

// ---------------- per-token attention (scores over heads, 8x8) ----------------
// One 256-thread block per token.
__global__ __launch_bounds__(256)
void attention_kernel(const bf16* __restrict__ qb, const bf16* __restrict__ kb,
                      const bf16* __restrict__ vb, bf16* __restrict__ ab)
{
    __shared__ float q[Oo];
    __shared__ float k[Oo];
    __shared__ float v[Oo];
    __shared__ float attn[Hh * Hh];

    const size_t base = (size_t)blockIdx.x * Oo;
    for (int i = threadIdx.x; i < Oo; i += 256) {
        q[i] = bf2f(qb[base + i]);
        k[i] = bf2f(kb[base + i]);
        v[i] = bf2f(vb[base + i]);
    }
    __syncthreads();

    // scores[h][t] = dot(q[h,:], k[t,:]) / sqrt(DH)
    if (threadIdx.x < Hh * Hh) {
        const int h = threadIdx.x >> 3;
        const int t = threadIdx.x & 7;
        const float* qh = q + h * DHd;
        const float* kt = k + t * DHd;
        float s = 0.f;
        for (int d = 0; d < DHd; d++) s += qh[d] * kt[d];
        attn[h * Hh + t] = s * (1.0f / 16.0f);   // 1/sqrt(256)
    }
    __syncthreads();

    // softmax over t (8 values per head)
    if (threadIdx.x < Hh) {
        const int h = threadIdx.x;
        float mx = attn[h * Hh];
        #pragma unroll
        for (int t = 1; t < Hh; t++) mx = fmaxf(mx, attn[h * Hh + t]);
        float e[Hh];
        float sum = 0.f;
        #pragma unroll
        for (int t = 0; t < Hh; t++) { e[t] = __expf(attn[h * Hh + t] - mx); sum += e[t]; }
        const float inv = 1.0f / sum;
        #pragma unroll
        for (int t = 0; t < Hh; t++) attn[h * Hh + t] = e[t] * inv;
    }
    __syncthreads();

    // attended[h][d] = sum_t attn[h][t] * v[t][d]
    const int d = threadIdx.x;               // 0..255 == DH
    #pragma unroll
    for (int h = 0; h < Hh; h++) {
        float s = 0.f;
        #pragma unroll
        for (int t = 0; t < Hh; t++) s += attn[h * Hh + t] * v[t * DHd + d];
        ab[base + h * DHd + d] = f2bf(s);
    }
}

// ---------------- in-place LayerNorm over C=2048 ----------------
__global__ __launch_bounds__(256)
void layernorm_kernel(float* __restrict__ y, const float* __restrict__ gamma,
                      const float* __restrict__ beta)
{
    const size_t base = (size_t)blockIdx.x * Cc;
    float vals[8];
    float s = 0.f, s2 = 0.f;
    #pragma unroll
    for (int p = 0; p < 8; p++) {
        float x = y[base + p * 256 + threadIdx.x];
        vals[p] = x;
        s += x;
        s2 += x * x;
    }
    __shared__ float red0[256];
    __shared__ float red1[256];
    red0[threadIdx.x] = s;
    red1[threadIdx.x] = s2;
    __syncthreads();
    for (int off = 128; off > 0; off >>= 1) {
        if (threadIdx.x < off) {
            red0[threadIdx.x] += red0[threadIdx.x + off];
            red1[threadIdx.x] += red1[threadIdx.x + off];
        }
        __syncthreads();
    }
    const float mu  = red0[0] * (1.0f / Cc);
    const float var = red1[0] * (1.0f / Cc) - mu * mu;
    const float inv = rsqrtf(var + EPSl);
    #pragma unroll
    for (int p = 0; p < 8; p++) {
        const int c = p * 256 + threadIdx.x;
        y[base + c] = gamma[c] * (vals[p] - mu) * inv + beta[c];
    }
}

// ---------------- host launcher ----------------
extern "C" void kernel_launch(void* const* d_in, const int* in_sizes, int n_in,
                              void* d_out, int out_size, void* d_ws, size_t ws_size,
                              hipStream_t stream)
{
    const float* x     = (const float*)d_in[0];
    const float* Wq    = (const float*)d_in[1];
    const float* bq    = (const float*)d_in[2];
    const float* Wk    = (const float*)d_in[3];
    const float* bk    = (const float*)d_in[4];
    const float* Wv    = (const float*)d_in[5];
    const float* bv    = (const float*)d_in[6];
    const float* Wo    = (const float*)d_in[7];
    const float* bo    = (const float*)d_in[8];
    const float* gamma = (const float*)d_in[9];
    const float* beta  = (const float*)d_in[10];
    float* out = (float*)d_out;

    char* ws = (char*)d_ws;
    size_t off = 0;
    auto alloc = [&](size_t bytes) -> void* {
        void* p = (void*)(ws + off);
        off = (off + bytes + 255) & ~(size_t)255;
        return p;
    };

    bf16* xb  = (bf16*)alloc((size_t)MTOK * Cc * sizeof(bf16));   // 64 MB
    bf16* Wtq = (bf16*)alloc((size_t)Oo * Cc * sizeof(bf16));     // 8 MB each
    bf16* Wtk = (bf16*)alloc((size_t)Oo * Cc * sizeof(bf16));
    bf16* Wtv = (bf16*)alloc((size_t)Oo * Cc * sizeof(bf16));
    bf16* Wto = (bf16*)alloc((size_t)Cc * Oo * sizeof(bf16));
    bf16* qb  = (bf16*)alloc((size_t)MTOK * Oo * sizeof(bf16));   // 64 MB each
    bf16* kb  = (bf16*)alloc((size_t)MTOK * Oo * sizeof(bf16));
    bf16* vb  = (bf16*)alloc((size_t)MTOK * Oo * sizeof(bf16));
    bf16* ab  = xb;  // x(bf16) is dead after QKV GEMMs; reuse for attended

    // 1) convert x to bf16
    const int NEL = MTOK * Cc;
    f32_to_bf16_kernel<<<NEL / 256, 256, 0, stream>>>(x, xb, NEL);

    // 2) transpose+convert weights: W[K,N] -> Wt[N,K]
    dim3 tb(32, 8);
    dim3 tg(Oo / 32, Cc / 32);
    transpose_f32_to_bf16<<<tg, tb, 0, stream>>>(Wq, Wtq, Cc, Oo);
    transpose_f32_to_bf16<<<tg, tb, 0, stream>>>(Wk, Wtk, Cc, Oo);
    transpose_f32_to_bf16<<<tg, tb, 0, stream>>>(Wv, Wtv, Cc, Oo);
    dim3 tgo(Cc / 32, Oo / 32);
    transpose_f32_to_bf16<<<tgo, tb, 0, stream>>>(Wo, Wto, Oo, Cc);

    // 3) QKV projections (WMMA bf16, f32 accumulate, bias fused)
    dim3 gb(128);
    dim3 gg(Oo / 128, MTOK / 128);   // (16, 128)
    gemm_wmma_bf16<0><<<gg, gb, 0, stream>>>(xb, Wtq, bq, nullptr, qb, nullptr, MTOK, Oo, Cc);
    gemm_wmma_bf16<0><<<gg, gb, 0, stream>>>(xb, Wtk, bk, nullptr, kb, nullptr, MTOK, Oo, Cc);
    gemm_wmma_bf16<0><<<gg, gb, 0, stream>>>(xb, Wtv, bv, nullptr, vb, nullptr, MTOK, Oo, Cc);

    // 4) per-token 8x8 head attention + softmax
    attention_kernel<<<MTOK, 256, 0, stream>>>(qb, kb, vb, ab);

    // 5) output projection with fused bias + residual -> d_out (f32)
    dim3 ggo(Cc / 128, MTOK / 128);
    gemm_wmma_bf16<1><<<ggo, gb, 0, stream>>>(ab, Wto, bo, x, nullptr, out, MTOK, Cc, Oo);

    // 6) in-place LayerNorm
    layernorm_kernel<<<MTOK, 256, 0, stream>>>(out, gamma, beta);
}
